// VGNet_36653250904398
// MI455X (gfx1250) — compile-verified, matrix operand order checked by reading, not run on hardware
//
#include <hip/hip_runtime.h>
#include <hip/hip_bf16.h>

typedef __attribute__((ext_vector_type(16))) _Float16 v16h;
typedef __attribute__((ext_vector_type(8)))  _Float16 v8h;
typedef __attribute__((ext_vector_type(8)))  float    v8f;

#define ALPHA    0.8f
#define BN_EPS   1e-5f
#define KPAD     256           // 245 padded to 256 for WMMA K
#define COUT1    128           // linear1 output channels
#define KSTEPS   (KPAD / 32)   // 8 WMMA K-steps
#define NTILES   (COUT1 / 16)  // 8 output-column tiles
#define MT       8             // M-tiles processed per wave

// ---------------------------------------------------------------------------
// generic zero fill
// ---------------------------------------------------------------------------
__global__ void zero_f32(float* __restrict__ p, long n) {
    long i = (long)blockIdx.x * blockDim.x + threadIdx.x;
    if (i < n) p[i] = 0.0f;
}

// ---------------------------------------------------------------------------
// CNN node encoder: conv1(3->8,3x3,s1)+relu, conv2(8->5,5x5,s2)+relu,
// flatten -> featH[N, 256] (f16, cols 245..255 zeroed).
// One block (128 threads) per node; everything staged in LDS.
// ---------------------------------------------------------------------------
__global__ void encoder_kernel(const float* __restrict__ x,
                               const float* __restrict__ c1w, const float* __restrict__ c1b,
                               const float* __restrict__ c2w, const float* __restrict__ c2b,
                               _Float16* __restrict__ featH, int N)
{
    __shared__ float xin[3 * 19 * 19];   // 1083
    __shared__ float h1s[8 * 17 * 17];   // 2312
    __shared__ float w1s[8 * 3 * 3 * 3]; // 216
    __shared__ float b1s[8];
    __shared__ float w2s[5 * 8 * 5 * 5]; // 1000
    __shared__ float b2s[8];

    const int node = blockIdx.x;
    const int tid  = threadIdx.x;
    if (node >= N) return;

    for (int i = tid; i < 1083; i += 128) xin[i] = x[(long)node * 1083 + i];
    for (int i = tid; i < 216;  i += 128) w1s[i] = c1w[i];
    for (int i = tid; i < 1000; i += 128) w2s[i] = c2w[i];
    if (tid < 8) b1s[tid] = c1b[tid];
    if (tid < 5) b2s[tid] = c2b[tid];
    __syncthreads();

    // conv1 + relu -> h1s [8,17,17]
    for (int o = tid; o < 8 * 17 * 17; o += 128) {
        int oc  = o / 289;
        int rem = o - oc * 289;
        int oy  = rem / 17;
        int ox  = rem - oy * 17;
        float s = b1s[oc];
        const float* wz = &w1s[oc * 27];
        #pragma unroll
        for (int ic = 0; ic < 3; ++ic)
            #pragma unroll
            for (int ky = 0; ky < 3; ++ky)
                #pragma unroll
                for (int kx = 0; kx < 3; ++kx)
                    s += xin[ic * 361 + (oy + ky) * 19 + (ox + kx)] * wz[ic * 9 + ky * 3 + kx];
        h1s[o] = s > 0.0f ? s : 0.0f;
    }
    __syncthreads();

    // conv2 + relu -> featH[node, 0..244]; pad 245..255 with zeros
    _Float16* frow = featH + (long)node * KPAD;
    for (int j = tid; j < 245; j += 128) {
        int oc  = j / 49;
        int rem = j - oc * 49;
        int oy  = rem / 7;
        int ox  = rem - oy * 7;
        int y0 = 2 * oy, x0 = 2 * ox;
        float s = b2s[oc];
        const float* wz = &w2s[oc * 200];
        for (int ic = 0; ic < 8; ++ic) {
            const float* hz = &h1s[ic * 289];
            #pragma unroll
            for (int ky = 0; ky < 5; ++ky)
                #pragma unroll
                for (int kx = 0; kx < 5; ++kx)
                    s += hz[(y0 + ky) * 17 + (x0 + kx)] * wz[ic * 25 + ky * 5 + kx];
        }
        s = s > 0.0f ? s : 0.0f;
        frow[j] = (_Float16)s;
    }
    for (int j = 245 + tid; j < KPAD; j += 128) frow[j] = (_Float16)0.0f;
}

// ---------------------------------------------------------------------------
// pack l1w [128,245] f32 -> WMMA B fragments, fragment-major:
// Bfrag[(((tileN*KSTEPS + s)*32 + lane)*16) + i] so each lane's 16 halves are
// contiguous (two b128 loads per fragment).
// ISA 16-bit B layout (32x16 KxN): lane<16 -> col=lane, K=s*32+{0..7,16..23};
// lane>=16 -> col=lane-16, K=s*32+{8..15,24..31}.
// ---------------------------------------------------------------------------
__global__ void pack_bfrag_kernel(const float* __restrict__ l1w, _Float16* __restrict__ Bfrag) {
    int idx = blockIdx.x * blockDim.x + threadIdx.x;
    if (idx >= NTILES * KSTEPS * 32 * 16) return;
    int i    = idx & 15;
    int lane = (idx >> 4) & 31;
    int s    = (idx >> 9) & (KSTEPS - 1);
    int tn   = idx >> 12;
    int col   = tn * 16 + (lane & 15);
    int khalf = (lane >> 4) * 8;
    int k = s * 32 + khalf + ((i < 8) ? i : (8 + i)); // i>=8: +16+(i-8)
    Bfrag[idx] = (k < 245) ? (_Float16)l1w[col * 245 + k] : (_Float16)0.0f;
}

// ---------------------------------------------------------------------------
// linear1 via WMMA: h0 = h = relu(featH @ l1w^T + l1b), f32 accumulate.
// One wave32 per (m-tile-group, n-tile). B fragments for the wave's n-tile are
// preloaded into 8 v16h registers and amortized over MT m-tiles; the inner
// loop is 2x global_load_b128 (A) + 1x v_wmma per K-step.
// ---------------------------------------------------------------------------
__global__ void linear1_wmma_kernel(const _Float16* __restrict__ featH, // [N, 256]
                                    const _Float16* __restrict__ Bfrag, // fragment-packed
                                    const float* __restrict__ l1b,
                                    float* __restrict__ h0, float* __restrict__ h, int N)
{
    const int tileN = blockIdx.y;
    const int lane  = threadIdx.x;        // 0..31
    const int khalf = (lane >> 4) * 8;    // 0 for lanes 0-15, 8 for lanes 16-31

    // preload all B fragments for this n-tile (loop-invariant across m-tiles)
    v16h breg[KSTEPS];
    const _Float16* bbase = Bfrag + ((long)tileN * KSTEPS * 32 + lane) * 16;
    #pragma unroll
    for (int s = 0; s < KSTEPS; ++s) {
        v8h blo = *(const v8h*)(bbase + (long)s * 32 * 16);
        v8h bhi = *(const v8h*)(bbase + (long)s * 32 * 16 + 8);
        #pragma unroll
        for (int i = 0; i < 8; ++i) { breg[s][i] = blo[i]; breg[s][8 + i] = bhi[i]; }
    }

    const int nAbs  = tileN * 16 + (lane & 15);
    const float bias = l1b[nAbs];
    const int moff  = (lane >> 4) ? 8 : 0;

    const int t0 = blockIdx.x * MT;
    for (int t = 0; t < MT; ++t) {
        const int tileM = t0 + t;
        if (tileM * 16 >= N) break;                 // wave-uniform
        int rowA = tileM * 16 + (lane & 15);
        if (rowA >= N) rowA = N - 1;                // clamp keeps EXEC all-1
        const _Float16* arow = featH + (long)rowA * KPAD;

        v8f acc = {};
        #pragma unroll
        for (int s = 0; s < KSTEPS; ++s) {
            // A fragment (16-bit A 16x32 layout): two contiguous 16B loads
            v8h alo = *(const v8h*)(arow + s * 32 + khalf);
            v8h ahi = *(const v8h*)(arow + s * 32 + khalf + 16);
            v16h a;
            #pragma unroll
            for (int i = 0; i < 8; ++i) { a[i] = alo[i]; a[8 + i] = ahi[i]; }
            acc = __builtin_amdgcn_wmma_f32_16x16x32_f16(
                      false, a, false, breg[s], (short)0, acc, false, false);
        }

        // C/D layout: lanes 0-15: VGPR r -> (M=r, N=lane); lanes 16-31: M=8+r
        const int mbase = tileM * 16 + moff;
        #pragma unroll
        for (int r = 0; r < 8; ++r) {
            int m = mbase + r;
            if (m < N) {
                float v = acc[r] + bias;
                v = v > 0.0f ? v : 0.0f;
                h0[(long)m * COUT1 + nAbs] = v;
                h [(long)m * COUT1 + nAbs] = v;
            }
        }
    }
}

// ---------------------------------------------------------------------------
// APPNP helpers
// ---------------------------------------------------------------------------
__global__ void deg_kernel(const long long* __restrict__ ei, const float* __restrict__ ea,
                           float* __restrict__ deg, int E) {
    int e = blockIdx.x * blockDim.x + threadIdx.x;
    if (e >= E) return;
    atomicAdd(&deg[(int)ei[(long)E + e]], ea[e]);   // col = edge_index[1]
}

__global__ void dinv_kernel(const float* __restrict__ deg, float* __restrict__ dinv, int N) {
    int i = blockIdx.x * blockDim.x + threadIdx.x;
    if (i >= N) return;
    float d = deg[i];
    dinv[i] = (d > 0.0f) ? rsqrtf(d) : 0.0f;
}

__global__ void norm_kernel(const long long* __restrict__ ei, const float* __restrict__ ea,
                            const float* __restrict__ dinv, float* __restrict__ norm, int E) {
    int e = blockIdx.x * blockDim.x + threadIdx.x;
    if (e >= E) return;
    int r = (int)ei[e];
    int c = (int)ei[(long)E + e];
    norm[e] = dinv[r] * ea[e] * dinv[c];
}

// one wave per edge: gather h[row] (128 ch), scatter-add into agg[col]
__global__ void prop_kernel(const long long* __restrict__ ei, const float* __restrict__ norm,
                            const float* __restrict__ h, float* __restrict__ agg, int E) {
    int e    = blockIdx.x * (blockDim.x >> 5) + (threadIdx.x >> 5);
    int lane = threadIdx.x & 31;
    if (e >= E) return;
    int r = (int)ei[e];
    int c = (int)ei[(long)E + e];
    float nw = norm[e];
    const float* hr = h   + (long)r * COUT1;
    float*       ac = agg + (long)c * COUT1;
    __builtin_prefetch(hr, 0, 0);                   // global_prefetch_b8
    #pragma unroll
    for (int j = 0; j < 4; ++j) {
        int ch = lane + 32 * j;
        atomicAdd(&ac[ch], nw * hr[ch]);
    }
}

__global__ void combine_kernel(const float* __restrict__ agg, const float* __restrict__ h0,
                               float* __restrict__ h, long n) {
    long i = (long)blockIdx.x * blockDim.x + threadIdx.x;
    if (i >= n) return;
    h[i] = (1.0f - ALPHA) * agg[i] + ALPHA * h0[i];
}

// ---------------------------------------------------------------------------
// BatchNorm stats: block-local accumulation, one atomic per block per channel
// ---------------------------------------------------------------------------
__global__ void bn_stats_kernel(const float* __restrict__ h, float* __restrict__ sum,
                                float* __restrict__ sumsq, int N) {
    const int c  = threadIdx.x;                 // 128 threads == channels
    const int r0 = blockIdx.x * 256;
    int r1 = r0 + 256; if (r1 > N) r1 = N;
    float s = 0.0f, ss = 0.0f;
    for (int r = r0; r < r1; ++r) {
        float v = h[(long)r * COUT1 + c];
        s += v; ss += v * v;
    }
    atomicAdd(&sum[c], s);
    atomicAdd(&sumsq[c], ss);
}

// BN apply + write node_x output + pooled sums
__global__ void bn_pool_kernel(const float* __restrict__ h, const float* __restrict__ sum,
                               const float* __restrict__ sumsq, const float* __restrict__ gamma,
                               const float* __restrict__ beta, const long long* __restrict__ batch,
                               float* __restrict__ node_out, float* __restrict__ gsum,
                               float* __restrict__ gcnt, int N) {
    long idx = (long)blockIdx.x * blockDim.x + threadIdx.x;
    if (idx >= (long)N * COUT1) return;
    int r = (int)(idx >> 7);
    int c = (int)(idx & 127);
    float invN = 1.0f / (float)N;
    float mean = sum[c] * invN;
    float var  = sumsq[c] * invN - mean * mean;
    float v = gamma[c] * (h[idx] - mean) * rsqrtf(var + BN_EPS) + beta[c];
    node_out[idx] = v;
    int g = (int)batch[r];
    atomicAdd(&gsum[(long)g * COUT1 + c], v);
    if (c == 0) atomicAdd(&gcnt[g], 1.0f);
}

__global__ void graphx_kernel(const float* __restrict__ gsum, const float* __restrict__ gcnt,
                              float* __restrict__ gx, int G) {
    int idx = blockIdx.x * blockDim.x + threadIdx.x;
    if (idx >= G * COUT1) return;
    int g = idx >> 7;
    float cnt = gcnt[g];
    gx[idx] = gsum[idx] / (cnt > 1.0f ? cnt : 1.0f);
}

// ---------------------------------------------------------------------------
// classifier MLP (tiny)
// ---------------------------------------------------------------------------
__global__ void mlp1_kernel(const float* __restrict__ gx, const float* __restrict__ l2w,
                            const float* __restrict__ l2b, float* __restrict__ hid, int G) {
    int idx = blockIdx.x * blockDim.x + threadIdx.x;
    if (idx >= G * 64) return;
    int g = idx / 64, j = idx - (idx / 64) * 64;
    float s = l2b[j];
    const float* gr = gx  + (long)g * 128;
    const float* wr = l2w + (long)j * 128;
    #pragma unroll 8
    for (int k = 0; k < 128; ++k) s += gr[k] * wr[k];
    hid[idx] = s > 0.0f ? s : 0.0f;
}

__global__ void mlp2_kernel(const float* __restrict__ hid, const float* __restrict__ l3w,
                            const float* __restrict__ l3b, float* __restrict__ pred, int G) {
    int idx = blockIdx.x * blockDim.x + threadIdx.x;
    if (idx >= G * 5) return;
    int g = idx / 5, j = idx - (idx / 5) * 5;
    float s = l3b[j];
    const float* hr = hid + (long)g * 64;
    const float* wr = l3w + (long)j * 64;
    #pragma unroll 8
    for (int k = 0; k < 64; ++k) s += hr[k] * wr[k];
    pred[idx] = s;
}

// ---------------------------------------------------------------------------
// host orchestration
// ---------------------------------------------------------------------------
static inline size_t alignup(size_t v) { return (v + 255) & ~(size_t)255; }

extern "C" void kernel_launch(void* const* d_in, const int* in_sizes, int n_in,
                              void* d_out, int out_size, void* d_ws, size_t ws_size,
                              hipStream_t stream) {
    const float*     x     = (const float*)d_in[0];
    const long long* ei    = (const long long*)d_in[1];
    const float*     ea    = (const float*)d_in[2];
    const long long* batch = (const long long*)d_in[3];
    const float* c1w = (const float*)d_in[5];
    const float* c1b = (const float*)d_in[6];
    const float* c2w = (const float*)d_in[7];
    const float* c2b = (const float*)d_in[8];
    const float* l1w = (const float*)d_in[9];
    const float* l1b = (const float*)d_in[10];
    const float* gam = (const float*)d_in[11];
    const float* bet = (const float*)d_in[12];
    const float* l2w = (const float*)d_in[13];
    const float* l2b = (const float*)d_in[14];
    const float* l3w = (const float*)d_in[15];
    const float* l3b = (const float*)d_in[16];
    (void)n_in; (void)ws_size;

    const int N = in_sizes[0] / (3 * 19 * 19);
    const int E = in_sizes[2];
    const int G = (out_size - N * COUT1) / 5;

    float* pred     = (float*)d_out;                 // [G,5]
    float* node_out = (float*)d_out + (long)G * 5;   // [N,128]

    // workspace carve-up
    char* ws = (char*)d_ws;
    size_t off = 0;
    auto take = [&](size_t bytes) { char* p = ws + off; off += alignup(bytes); return p; };
    _Float16* featH = (_Float16*)take((size_t)N * KPAD * sizeof(_Float16));
    _Float16* Bfrag = (_Float16*)take((size_t)NTILES * KSTEPS * 32 * 16 * sizeof(_Float16));
    float* h0    = (float*)take((size_t)N * COUT1 * sizeof(float));
    float* h     = (float*)take((size_t)N * COUT1 * sizeof(float));
    float* agg   = (float*)take((size_t)N * COUT1 * sizeof(float));
    float* deg   = (float*)take((size_t)N * sizeof(float));
    float* dinv  = (float*)take((size_t)N * sizeof(float));
    float* norm  = (float*)take((size_t)E * sizeof(float));
    float* bsum  = (float*)take(COUT1 * sizeof(float));
    float* bsq   = (float*)take(COUT1 * sizeof(float));
    float* gsum  = (float*)take((size_t)G * COUT1 * sizeof(float));
    float* gcnt  = (float*)take((size_t)G * sizeof(float));
    float* gx    = (float*)take((size_t)G * COUT1 * sizeof(float));
    float* hid   = (float*)take((size_t)G * 64 * sizeof(float));

    auto blocks = [](long n, int t) { return (unsigned)((n + t - 1) / t); };

    // --- encoder + linear1 (WMMA) ---
    pack_bfrag_kernel<<<blocks(NTILES * KSTEPS * 32 * 16, 256), 256, 0, stream>>>(l1w, Bfrag);
    encoder_kernel<<<N, 128, 0, stream>>>(x, c1w, c1b, c2w, c2b, featH, N);
    const int mtiles = (N + 15) / 16;
    dim3 g1((mtiles + MT - 1) / MT, NTILES);
    linear1_wmma_kernel<<<g1, 32, 0, stream>>>(featH, Bfrag, l1b, h0, h, N);

    // --- APPNP normalization ---
    zero_f32<<<blocks(N, 256), 256, 0, stream>>>(deg, N);
    deg_kernel<<<blocks(E, 256), 256, 0, stream>>>(ei, ea, deg, E);
    dinv_kernel<<<blocks(N, 256), 256, 0, stream>>>(deg, dinv, N);
    norm_kernel<<<blocks(E, 256), 256, 0, stream>>>(ei, ea, dinv, norm, E);

    // --- 2 propagation hops ---
    const long NC = (long)N * COUT1;
    for (int hop = 0; hop < 2; ++hop) {
        zero_f32<<<blocks(NC, 256), 256, 0, stream>>>(agg, NC);
        prop_kernel<<<blocks((long)E * 32, 256), 256, 0, stream>>>(ei, norm, h, agg, E);
        combine_kernel<<<blocks(NC, 256), 256, 0, stream>>>(agg, h0, h, NC);
    }

    // --- BatchNorm + node output + mean pool ---
    zero_f32<<<1, 256, 0, stream>>>(bsum, COUT1);
    zero_f32<<<1, 256, 0, stream>>>(bsq, COUT1);
    bn_stats_kernel<<<blocks(N, 256), 128, 0, stream>>>(h, bsum, bsq, N);
    zero_f32<<<blocks((long)G * COUT1, 256), 256, 0, stream>>>(gsum, (long)G * COUT1);
    zero_f32<<<blocks(G, 256), 256, 0, stream>>>(gcnt, G);
    bn_pool_kernel<<<blocks(NC, 256), 256, 0, stream>>>(h, bsum, bsq, gam, bet, batch,
                                                        node_out, gsum, gcnt, N);
    graphx_kernel<<<blocks((long)G * COUT1, 256), 256, 0, stream>>>(gsum, gcnt, gx, G);

    // --- classifier ---
    mlp1_kernel<<<blocks((long)G * 64, 256), 256, 0, stream>>>(gx, l2w, l2b, hid, G);
    mlp2_kernel<<<blocks((long)G * 5, 256), 256, 0, stream>>>(hid, l3w, l3b, pred, G);
}
